// DynamicEdge_54357106098789
// MI455X (gfx1250) — compile-verified, matrix operand order checked by reading, not run on hardware
//
#include <hip/hip_runtime.h>
#include <hip/hip_bf16.h>

// DGCNN (2x DynamicEdgeConv + linear head) for MI455X / gfx1250.
// All matmul work (kNN Gram matrix + EdgeConv MLPs) runs on
// v_wmma_f32_16x16x32_f16; graph features are staged in the 320KB WGP LDS.

typedef __attribute__((ext_vector_type(16))) _Float16 v16h;
typedef __attribute__((ext_vector_type(8)))  _Float16 v8h;
typedef __attribute__((ext_vector_type(8)))  float    v8f;
typedef __attribute__((ext_vector_type(4)))  float    v4f;

#define Bg 32
#define Np 2048
#define Kn 6
#define C1 16
#define Hd 32
#define DTS 20  // distance-tile row stride (floats): 16B aligned, conflict-free

static __device__ __forceinline__ v8f wmma_f16(v16h a, v16h b, v8f c) {
  // D = A(16x32 f16) * B(32x16 f16) + C(16x16 f32)
  return __builtin_amdgcn_wmma_f32_16x16x32_f16(
      /*neg_a=*/false, a, /*neg_b=*/false, b,
      /*c_mod=*/(short)0, c, /*reuse_a=*/false, /*reuse_b=*/false);
}

union FragU { v16h v; v8h h[2]; };

// ---------------------------------------------------------------------------
// kNN: one workgroup = (graph b, 16-row tile). Whole graph staged in LDS as
// f16 padded to 32 channels; Gram tiles via WMMA; per-lane top-6 + LDS merge.
// ---------------------------------------------------------------------------
template <int C>
__global__ __launch_bounds__(256) void knn_kernel(const float* __restrict__ x,
                                                  int* __restrict__ idx_out) {
  extern __shared__ __align__(16) char smem[];
  _Float16* __restrict__ Xs = (_Float16*)smem;                  // [Np][32] f16
  float* __restrict__ sq    = (float*)(smem + (size_t)Np * 32 * 2);   // [Np]
  float* __restrict__ dtile = (float*)(smem + (size_t)Np * 32 * 2 + Np * 4);
  float* __restrict__ mtv   = dtile + 8 * 16 * DTS;             // [16][16][6]
  int*   __restrict__ mti   = (int*)(mtv + 16 * 16 * 6);        // [16][16][6]

  const int tid  = threadIdx.x;
  const int lane = tid & 31;
  const int wave = tid >> 5;
  const int half = lane >> 4;  // 0: lanes 0-15, 1: lanes 16-31
  const int l15  = lane & 15;
  const int b    = blockIdx.x / (Np / 16);
  const int rt   = blockIdx.x % (Np / 16);
  const int rowBase = rt * 16;
  const float* xb = x + (size_t)b * Np * C;

  // Stage features (zero-pad channels to 32) + squared norms, vectorized:
  // float4 global loads -> packed v8h -> ds_store_b128.
  for (int p = tid; p < Np; p += 256) {
    float s = 0.f;
#pragma unroll
    for (int c8 = 0; c8 < C / 8; ++c8) {
      const v4f a = *(const v4f*)(xb + (size_t)p * C + c8 * 8);
      const v4f d = *(const v4f*)(xb + (size_t)p * C + c8 * 8 + 4);
      v8h o;
#pragma unroll
      for (int e = 0; e < 4; ++e) {
        s += a[e] * a[e] + d[e] * d[e];
        o[e]     = (_Float16)a[e];
        o[4 + e] = (_Float16)d[e];
      }
      *(v8h*)(Xs + p * 32 + c8 * 8) = o;
    }
    if (C == 16) {  // zero-pad channels 16..31
      const v8h z = {};
      *(v8h*)(Xs + p * 32 + 16) = z;
      *(v8h*)(Xs + p * 32 + 24) = z;
    }
    sq[p] = s;
  }
  __syncthreads();

  // A fragment (this wave's 16 rows) - ISA 16-bit A layout:
  // lanes 0-15 hold K 0-7 & 16-23, lanes 16-31 hold K 8-15 & 24-31.
  const int c0 = half * 8;
  FragU ua;
  {
    const int p = rowBase + l15;
    ua.h[0] = *(const v8h*)(Xs + p * 32 + c0);
    ua.h[1] = *(const v8h*)(Xs + p * 32 + c0 + 16);
  }

  // Hoist loop-invariant row norms: sq[rowBase + half*8 + 0..7].
  const v4f sr0 = *(const v4f*)(sq + rowBase + half * 8);
  const v4f sr1 = *(const v4f*)(sq + rowBase + half * 8 + 4);
  float sqr[8];
#pragma unroll
  for (int v = 0; v < 4; ++v) { sqr[v] = sr0[v]; sqr[4 + v] = sr1[v]; }

  float tv[6]; int ti[6];
#pragma unroll
  for (int s = 0; s < 6; ++s) { tv[s] = 3.0e38f; ti[s] = 0; }

  float* __restrict__ dt = dtile + wave * 16 * DTS;

  // Software pipeline: preload first B fragment.
  FragU ub;
  {
    const int p = wave * 16 + l15;
    ub.h[0] = *(const v8h*)(Xs + p * 32 + c0);
    ub.h[1] = *(const v8h*)(Xs + p * 32 + c0 + 16);
  }

  for (int t = wave; t < Np / 16; t += 8) {
    const int colBase = t * 16;
    v8f acc = {};
    acc = wmma_f16(ua.v, ub.v, acc);

    // Unconditional (wrapped-index) preload of the next tile's B fragment:
    // no exec-mask divergence; drains while the epilogue runs.
    {
      const int tn = (t + 8 < Np / 16) ? (t + 8) : wave;
      const int p  = tn * 16 + l15;
      ub.h[0] = *(const v8h*)(Xs + p * 32 + c0);
      ub.h[1] = *(const v8h*)(Xs + p * 32 + c0 + 16);
    }

    const float sqc = sq[colBase + l15];
    if (t == rt) {
      // Self column lives only in this tile (uniform branch, 1/16 iters).
#pragma unroll
      for (int v = 0; v < 8; ++v) {
        float d = sqr[v] + sqc - 2.0f * acc[v];
        if (v + half * 8 == l15) d = 3.0e38f;  // exclude self
        dt[(v + half * 8) * DTS + l15] = d;
      }
    } else {
#pragma unroll
      for (int v = 0; v < 8; ++v) {
        float d = sqr[v] + sqc - 2.0f * acc[v];
        dt[(v + half * 8) * DTS + l15] = d;
      }
    }
    // Per-lane scan: row l15, columns half*8 .. half*8+7, as 2x b128 loads
    // (wave-private LDS region; DS ops from one wave stay in order).
    const v4f ca = *(const v4f*)(dt + l15 * DTS + half * 8);
    const v4f cb = *(const v4f*)(dt + l15 * DTS + half * 8 + 4);
#pragma unroll
    for (int j = 0; j < 8; ++j) {
      const float d  = (j < 4) ? ca[j & 3] : cb[j & 3];
      const int   ci = colBase + half * 8 + j;
      if (d < tv[5]) {
        tv[5] = d; ti[5] = ci;
#pragma unroll
        for (int s = 5; s > 0; --s) {
          if (tv[s] < tv[s - 1]) {
            float tf = tv[s]; tv[s] = tv[s - 1]; tv[s - 1] = tf;
            int   tt = ti[s]; ti[s] = ti[s - 1]; ti[s - 1] = tt;
          }
        }
      }
    }
  }

  // Merge 16 partial lists per row.
  const int part = wave * 2 + half;
#pragma unroll
  for (int s = 0; s < 6; ++s) {
    mtv[(l15 * 16 + part) * 6 + s] = tv[s];
    mti[(l15 * 16 + part) * 6 + s] = ti[s];
  }
  __syncthreads();
  if (tid < 16) {
    float ftv[6]; int fti[6];
#pragma unroll
    for (int s = 0; s < 6; ++s) { ftv[s] = 3.0e38f; fti[s] = 0; }
    for (int pp = 0; pp < 16; ++pp) {
      for (int s = 0; s < 6; ++s) {
        float d  = mtv[(tid * 16 + pp) * 6 + s];
        int   ci = mti[(tid * 16 + pp) * 6 + s];
        if (d < ftv[5]) {
          ftv[5] = d; fti[5] = ci;
#pragma unroll
          for (int u = 5; u > 0; --u) {
            if (ftv[u] < ftv[u - 1]) {
              float tf = ftv[u]; ftv[u] = ftv[u - 1]; ftv[u - 1] = tf;
              int   tt = fti[u]; fti[u] = fti[u - 1]; fti[u - 1] = tt;
            }
          }
        }
      }
    }
    const int grow = b * Np + rowBase + tid;
#pragma unroll
    for (int s = 0; s < 6; ++s) idx_out[grow * Kn + s] = b * Np + fti[s];
  }
}

// ---------------------------------------------------------------------------
// EdgeConv: wave handles 16 points; loop over 6 neighbors; 2-layer MLP on
// [x_i, x_j - x_i] via WMMA; k-max folded into the accumulator registers.
// ---------------------------------------------------------------------------
template <int CIN>
__global__ __launch_bounds__(256) void edge_conv_kernel(
    const float* __restrict__ xin, const int* __restrict__ idx,
    const float* __restrict__ Wa, const float* __restrict__ ba,
    const float* __restrict__ Wb, const float* __restrict__ bb,
    float* __restrict__ hout) {
  constexpr int K1  = 2 * CIN;   // layer-1 K (32 or 64)
  constexpr int KCH = K1 / 32;   // K chunks of 32
  __shared__ __align__(16) _Float16 sWa[K1 * Hd];
  __shared__ __align__(16) _Float16 sWb[Hd * Hd];
  __shared__ float sba[Hd], sbb[Hd];
  __shared__ __align__(16) _Float16 sE[8][16 * K1];  // per-wave edge tiles
  __shared__ __align__(16) _Float16 sH[8][16 * Hd];  // per-wave hidden tiles

  const int tid = threadIdx.x, lane = tid & 31, wave = tid >> 5;
  const int half = lane >> 4, l15 = lane & 15;

  for (int i = tid; i < K1 * Hd; i += 256) sWa[i] = (_Float16)Wa[i];
  for (int i = tid; i < Hd * Hd; i += 256) sWb[i] = (_Float16)Wb[i];
  if (tid < Hd) { sba[tid] = ba[tid]; sbb[tid] = bb[tid]; }
  __syncthreads();

  // Weight B-fragments (strided LDS gathers, done once, kept in VGPRs).
  v16h fWa[KCH][2], fWb[2];
#pragma unroll
  for (int kc = 0; kc < KCH; ++kc) {
#pragma unroll
    for (int nt = 0; nt < 2; ++nt) {
      const int col = nt * 16 + l15;
      const int kb  = kc * 32 + half * 8;
      v16h f;
#pragma unroll
      for (int e = 0; e < 8; ++e) {
        f[e]     = sWa[(kb + e) * Hd + col];
        f[8 + e] = sWa[(kb + 16 + e) * Hd + col];
      }
      fWa[kc][nt] = f;
    }
  }
#pragma unroll
  for (int nt = 0; nt < 2; ++nt) {
    const int col = nt * 16 + l15;
    const int kb  = half * 8;
    v16h f;
#pragma unroll
    for (int e = 0; e < 8; ++e) {
      f[e]     = sWb[(kb + e) * Hd + col];
      f[8 + e] = sWb[(kb + 16 + e) * Hd + col];
    }
    fWb[nt] = f;
  }

  const int   ptBase = (blockIdx.x * 8 + wave) * 16;
  const float b1_0 = sba[l15],      b1_1 = sba[16 + l15];
  const float b2_0 = sbb[l15],      b2_1 = sbb[16 + l15];

  v8f m0, m1;
#pragma unroll
  for (int v = 0; v < 8; ++v) { m0[v] = -3.0e38f; m1[v] = -3.0e38f; }

  _Float16* __restrict__ eT = sE[wave];
  _Float16* __restrict__ hT = sH[wave];

  const int pg = ptBase + l15;
  const float* xi = xin + (size_t)pg * CIN;

  for (int k = 0; k < Kn; ++k) {
    // Build edge tile: lanes 0-15 write x_i, lanes 16-31 write x_j - x_i.
    // float4 global loads -> packed v8h -> ds_store_b128.
    if (half == 0) {
#pragma unroll
      for (int c8 = 0; c8 < CIN / 8; ++c8) {
        const v4f a = *(const v4f*)(xi + c8 * 8);
        const v4f b = *(const v4f*)(xi + c8 * 8 + 4);
        v8h o;
#pragma unroll
        for (int e = 0; e < 4; ++e) { o[e] = (_Float16)a[e]; o[4 + e] = (_Float16)b[e]; }
        *(v8h*)(eT + l15 * K1 + c8 * 8) = o;
      }
    } else {
      const int ng = idx[pg * Kn + k];
      const float* xj = xin + (size_t)ng * CIN;
#pragma unroll
      for (int c8 = 0; c8 < CIN / 8; ++c8) {
        const v4f a0 = *(const v4f*)(xi + c8 * 8);
        const v4f a1 = *(const v4f*)(xi + c8 * 8 + 4);
        const v4f j0 = *(const v4f*)(xj + c8 * 8);
        const v4f j1 = *(const v4f*)(xj + c8 * 8 + 4);
        v8h o;
#pragma unroll
        for (int e = 0; e < 4; ++e) {
          o[e]     = (_Float16)(j0[e] - a0[e]);
          o[4 + e] = (_Float16)(j1[e] - a1[e]);
        }
        *(v8h*)(eT + l15 * K1 + CIN + c8 * 8) = o;
      }
    }
    // A fragments from the edge tile.
    FragU ea[KCH];
#pragma unroll
    for (int kc = 0; kc < KCH; ++kc) {
      const int c0 = kc * 32 + half * 8;
      ea[kc].h[0] = *(const v8h*)(eT + l15 * K1 + c0);
      ea[kc].h[1] = *(const v8h*)(eT + l15 * K1 + c0 + 16);
    }
    // Layer 1 (+bias, ReLU) -> f16 LDS tile for layer 2's A fragment.
#pragma unroll
    for (int nt = 0; nt < 2; ++nt) {
      v8f a = {};
#pragma unroll
      for (int kc = 0; kc < KCH; ++kc) a = wmma_f16(ea[kc].v, fWa[kc][nt], a);
      const float bv = nt ? b1_1 : b1_0;
#pragma unroll
      for (int v = 0; v < 8; ++v) {
        float h = a[v] + bv;
        h = h > 0.f ? h : 0.f;
        hT[(v + half * 8) * Hd + nt * 16 + l15] = (_Float16)h;
      }
    }
    FragU ha;
    ha.h[0] = *(const v8h*)(hT + l15 * Hd + half * 8);
    ha.h[1] = *(const v8h*)(hT + l15 * Hd + half * 8 + 16);
    // Layer 2, running max over neighbors directly in accumulators.
#pragma unroll
    for (int nt = 0; nt < 2; ++nt) {
      v8f a = {};
      a = wmma_f16(ha.v, fWb[nt], a);
      const float bv = nt ? b2_1 : b2_0;
      v8f& m = nt ? m1 : m0;
#pragma unroll
      for (int v = 0; v < 8; ++v) {
        float h = a[v] + bv;
        m[v] = m[v] > h ? m[v] : h;
      }
    }
  }
#pragma unroll
  for (int nt = 0; nt < 2; ++nt) {
    const v8f& m = nt ? m1 : m0;
#pragma unroll
    for (int v = 0; v < 8; ++v)
      hout[(size_t)(ptBase + v + half * 8) * Hd + nt * 16 + l15] = m[v];
  }
}

// ---------------------------------------------------------------------------
// Head: relu(h2 @ Wl1 + bl1) @ Wl2 + bl2   (tiny, VALU)
// ---------------------------------------------------------------------------
__global__ __launch_bounds__(256) void head_kernel(
    const float* __restrict__ h, const float* __restrict__ Wl1,
    const float* __restrict__ bl1, const float* __restrict__ Wl2,
    const float* __restrict__ bl2, float* __restrict__ out) {
  __shared__ float sW1[32 * 16], sb1[16], sW2[16 * 4], sb2[4];
  const int tid = threadIdx.x;
  for (int i = tid; i < 32 * 16; i += 256) sW1[i] = Wl1[i];
  if (tid < 16) sb1[tid] = bl1[tid];
  if (tid < 64) sW2[tid] = Wl2[tid];
  if (tid < 4)  sb2[tid] = bl2[tid];
  __syncthreads();
  const int p = blockIdx.x * 256 + tid;
  float hv[32];
#pragma unroll
  for (int c8 = 0; c8 < 8; ++c8) {
    const v4f a = *(const v4f*)(h + (size_t)p * 32 + c8 * 4);
#pragma unroll
    for (int e = 0; e < 4; ++e) hv[c8 * 4 + e] = a[e];
  }
  float t[16];
#pragma unroll
  for (int o = 0; o < 16; ++o) {
    float s = sb1[o];
#pragma unroll
    for (int c = 0; c < 32; ++c) s += hv[c] * sW1[c * 16 + o];
    t[o] = s > 0.f ? s : 0.f;
  }
#pragma unroll
  for (int o = 0; o < 4; ++o) {
    float s = sb2[o];
#pragma unroll
    for (int c = 0; c < 16; ++c) s += t[c] * sW2[c * 4 + o];
    out[(size_t)p * 4 + o] = s;
  }
}

// ---------------------------------------------------------------------------
extern "C" void kernel_launch(void* const* d_in, const int* in_sizes, int n_in,
                              void* d_out, int out_size, void* d_ws,
                              size_t ws_size, hipStream_t stream) {
  const float* x   = (const float*)d_in[0];
  // d_in[1] = batch ids: contiguous equal-sized graphs -> implied by layout.
  const float* W1  = (const float*)d_in[2];
  const float* b1  = (const float*)d_in[3];
  const float* W2  = (const float*)d_in[4];
  const float* b2  = (const float*)d_in[5];
  const float* W3  = (const float*)d_in[6];
  const float* b3  = (const float*)d_in[7];
  const float* W4  = (const float*)d_in[8];
  const float* b4  = (const float*)d_in[9];
  const float* Wl1 = (const float*)d_in[10];
  const float* bl1 = (const float*)d_in[11];
  const float* Wl2 = (const float*)d_in[12];
  const float* bl2 = (const float*)d_in[13];
  float* outp = (float*)d_out;

  // Workspace layout: h1 (8MB) | h2 (8MB) | idx (1.5MB)
  float* h1  = (float*)d_ws;
  float* h2  = h1 + (size_t)Bg * Np * Hd;
  int*   idx = (int*)(h2 + (size_t)Bg * Np * Hd);

  const size_t KNN_SMEM = (size_t)Np * 32 * 2   // Xs f16 [2048][32]
                        + (size_t)Np * 4        // sq
                        + 8 * 16 * DTS * 4      // dtile
                        + 16 * 16 * 6 * 4 * 2;  // merge lists (f32 + i32)

  (void)hipFuncSetAttribute((const void*)knn_kernel<C1>,
                            hipFuncAttributeMaxDynamicSharedMemorySize,
                            (int)KNN_SMEM);
  (void)hipFuncSetAttribute((const void*)knn_kernel<Hd>,
                            hipFuncAttributeMaxDynamicSharedMemorySize,
                            (int)KNN_SMEM);

  const int knnGrid  = Bg * (Np / 16);       // 4096 WGs
  const int convGrid = (Bg * Np) / 128;      // 512 WGs (8 waves x 16 points)

  knn_kernel<C1><<<knnGrid, 256, KNN_SMEM, stream>>>(x, idx);
  edge_conv_kernel<C1><<<convGrid, 256, 0, stream>>>(x, idx, W1, b1, W2, b2, h1);
  knn_kernel<Hd><<<knnGrid, 256, KNN_SMEM, stream>>>(h1, idx);
  edge_conv_kernel<Hd><<<convGrid, 256, 0, stream>>>(h1, idx, W3, b3, W4, b4, h2);
  head_kernel<<<(Bg * Np) / 256, 256, 0, stream>>>(h2, Wl1, bl1, Wl2, bl2, outp);
}
